// ResGraphConvUnpool_57964878627090
// MI455X (gfx1250) — matrix operand order repaired and checked by reading, not compiled, not run on hardware
//
#include <hip/hip_runtime.h>

// ---------------------------------------------------------------------------
// ResGraphConvUnpool for MI455X (gfx1250): bf16 WMMA for the kNN Gram matrix
// and all 128x128 GEMMs; TDM (tensor_load_to_lds) with LDS padding for tile
// staging; GCN aggregation exploits deg==9 (constant).
// Workspace requirement: ~30.8 MB (see layout in kernel_launch).
// ---------------------------------------------------------------------------

#define N_PTS   4096
#define DIMF    128
#define KNN     8
#define NBLK    12
#define BATCH   4
#define MROWS   (BATCH * N_PTS)   // 16384
#define LDST    136               // padded LDS row stride in bf16 elems (272B)
#define DSTR    68                // padded LDS column stride for gram (f32)

typedef __attribute__((ext_vector_type(16))) __bf16 v16bf;
typedef __attribute__((ext_vector_type(8)))  float  v8f;

#if defined(__has_builtin)
#if __has_builtin(__builtin_amdgcn_tensor_load_to_lds) && \
    __has_builtin(__builtin_amdgcn_s_wait_tensorcnt)
#define HAVE_TDM 1
#endif
#endif
#ifndef HAVE_TDM
#define HAVE_TDM 0
#endif

#if HAVE_TDM
typedef __attribute__((ext_vector_type(4))) unsigned int v4u;
typedef __attribute__((ext_vector_type(8))) int          v8i;
typedef __attribute__((ext_vector_type(4))) int          v4i;

// TDM load of a (rows x 128) bf16 tile into LDS with row padding 256B -> 272B
// (pad_interval: 64 DWORDs, pad_amount: 4 DWORDs), i.e. LDS stride LDST=136.
__device__ __forceinline__ void tdm_load_bf16_tile(unsigned lds_off,
                                                   unsigned long long ga,
                                                   int rows) {
  v4u g0 = {1u,                                   // count=1 (user descriptor)
            lds_off,                              // lds_addr
            (unsigned)(ga & 0xFFFFFFFFull),       // global_addr[31:0]
            (unsigned)((ga >> 32) & 0x1FFFFFFull) | (2u << 30)}; // [56:32]|type=2
  v8i g1;
  g1[0] = (int)((1u << 16)      // data_size = 2 bytes
              | (1u << 20)      // pad_enable
              | (5u << 22)      // pad_interval = 64 DWORDs (256B)
              | (3u << 25));    // pad_amount   = 4 DWORDs (16B)
  g1[1] = (int)(128u << 16);                 // tensor_dim0 = 128 (lo bits)
  g1[2] = (int)((unsigned)rows << 16);       // dim0 hi=0 | tensor_dim1 lo=rows
  g1[3] = (int)(128u << 16);                 // dim1 hi=0 | tile_dim0 = 128
  g1[4] = rows;                              // tile_dim1 = rows | tile_dim2=0
  g1[5] = 128;                               // tensor_dim0_stride = 128 elems
  g1[6] = 0;
  g1[7] = 0;
  v4i gz4 = {0, 0, 0, 0};
  v8i gz8 = {0, 0, 0, 0, 0, 0, 0, 0};
  __builtin_amdgcn_tensor_load_to_lds(g0, g1, gz4, gz4, gz8, 0);
}
#endif

__device__ __forceinline__ unsigned short f2bf(float x) {
  unsigned int u = __float_as_uint(x);
  u += 0x7FFFu + ((u >> 16) & 1u);   // round-to-nearest-even
  return (unsigned short)(u >> 16);
}

// A operand (16x32 bf16, MxK): lane m=lane&15, half h=lane>>4.
// elems 0..7  = row[k0 + 8h .. +7], elems 8..15 = row[k0 + 16 + 8h .. +7]
__device__ __forceinline__ v16bf load_a(const unsigned short* row, int k0, int half) {
  union { v16bf v; uint4 u[2]; } t;
  t.u[0] = *(const uint4*)(row + k0 + half * 8);
  t.u[1] = *(const uint4*)(row + k0 + 16 + half * 8);
  return t.v;
}
// B operand (32x16 bf16, KxN): lane n=lane&15, half h=lane>>4 -> elems 0..7 =
// K k0 + 8h .. +7, elems 8..15 = K k0 + 16 + 8h .. +7 of column n's feature row.
__device__ __forceinline__ v16bf load_b(const unsigned short* row, int k0, int half) {
  union { v16bf v; uint4 u[2]; } t;
  t.u[0] = *(const uint4*)(row + k0 + half * 8);
  t.u[1] = *(const uint4*)(row + k0 + 16 + half * 8);
  return t.v;
}

// ---------------------------------------------------------------------------
// features (B,DIM,N) -> h (B*N, DIM)
__global__ __launch_bounds__(256) void t_in_kernel(const float* __restrict__ f,
                                                   float* __restrict__ h) {
  int i = blockIdx.x * 256 + threadIdx.x;          // over B*DIM*N = 2097152
  int n = i & 4095;
  int d = (i >> 12) & 127;
  int b = i >> 19;
  h[((b << 12) + n) * DIMF + d] = f[i];
}

// h (B*N, DIM) -> out (B,DIM,N)
__global__ __launch_bounds__(256) void t_out_kernel(const float* __restrict__ h,
                                                    float* __restrict__ o) {
  int i = blockIdx.x * 256 + threadIdx.x;
  int n = i & 4095;
  int d = (i >> 12) & 127;
  int b = i >> 19;
  o[i] = h[((b << 12) + n) * DIMF + d];
}

// Convert Wa, Wb (12,128,128)[k][n] and Wp1 (128,128)[k][n] to bf16 [n][k].
__global__ __launch_bounds__(256) void conv_w_kernel(const float* __restrict__ Wa,
                                                     const float* __restrict__ Wb,
                                                     const float* __restrict__ Wp1,
                                                     unsigned short* __restrict__ Wta,
                                                     unsigned short* __restrict__ Wtb,
                                                     unsigned short* __restrict__ Wtp1) {
  int i = blockIdx.x * 256 + threadIdx.x;
  if (i < NBLK * DIMF * DIMF) {
    int blk = i >> 14, r = i & 16383;
    int n = r >> 7, k = r & 127;
    Wta[i] = f2bf(Wa[(blk << 14) + k * DIMF + n]);
    Wtb[i] = f2bf(Wb[(blk << 14) + k * DIMF + n]);
  } else if (i < NBLK * DIMF * DIMF + DIMF * DIMF) {
    int r = i - NBLK * DIMF * DIMF;
    int n = r >> 7, k = r & 127;
    Wtp1[r] = f2bf(Wp1[k * DIMF + n]);
  }
}

// Per-block prep: bf16 copy of h + squared norms. One wave per row.
__global__ __launch_bounds__(256) void prep_kernel(const float* __restrict__ h,
                                                   unsigned short* __restrict__ hbf,
                                                   float* __restrict__ sq) {
  int m = blockIdx.x * 8 + (threadIdx.x >> 5);
  int lane = threadIdx.x & 31;
  float4 v = ((const float4*)(h + (size_t)m * DIMF))[lane];
  uint2 pk;
  pk.x = (unsigned)f2bf(v.x) | ((unsigned)f2bf(v.y) << 16);
  pk.y = (unsigned)f2bf(v.z) | ((unsigned)f2bf(v.w) << 16);
  ((uint2*)(hbf + (size_t)m * DIMF))[lane] = pk;
  float s = v.x * v.x + v.y * v.y + v.z * v.z + v.w * v.w;
  for (int off = 16; off; off >>= 1) s += __shfl_xor(s, off, 32);
  if (lane == 0) sq[m] = s;
}

// ---------------------------------------------------------------------------
// kNN: each workgroup handles 64 query rows of one batch. Streams 32 chunks of
// 128 candidates through LDS (TDM), Gram via WMMA bf16 into a padded
// column-major LDS tile, then 64 threads keep a running top-8.
// LDS: sQ 64x136 bf16 | sC 128x136 bf16 | sD 128x68 f32 (col-major) | cs 128
__global__ __launch_bounds__(256) void knn_kernel(const unsigned short* __restrict__ hbf,
                                                  const float* __restrict__ sq,
                                                  int* __restrict__ idx) {
  extern __shared__ __align__(16) char smem[];
  unsigned short* sQ = (unsigned short*)smem;
  unsigned short* sC = sQ + 64 * LDST;
  float* sD = (float*)(sC + 128 * LDST);            // [col*DSTR + row]
  float* cs = sD + 128 * DSTR;

  int b = blockIdx.x >> 6;
  int rb = blockIdx.x & 63;
  int qlocal0 = rb * 64;
  int qglob0 = b * N_PTS + qlocal0;
  int tid = threadIdx.x, lane = tid & 31, wave = tid >> 5;
  int half = lane >> 4, m_in = lane & 15;

#if HAVE_TDM
  if (wave == 0) {
    tdm_load_bf16_tile((unsigned)(unsigned long long)(void*)sQ,
                       (unsigned long long)(const void*)(hbf + (size_t)qglob0 * DIMF),
                       64);
    __builtin_amdgcn_s_wait_tensorcnt(0);
  }
#else
  {
    const uint4* gq = (const uint4*)(hbf + (size_t)qglob0 * DIMF);
    for (int i = tid; i < 64 * 16; i += 256) {
      int r = i >> 4, ch = i & 15;
      *(uint4*)(sQ + r * LDST + ch * 8) = gq[i];
    }
  }
#endif
  __syncthreads();

  // hoist A operands: fixed query rows per wave for the whole kernel
  int rt = wave >> 1;
  const unsigned short* arow = &sQ[(rt * 16 + m_in) * LDST];
  v16bf av[4];
#pragma unroll
  for (int kk = 0; kk < 4; kk++) av[kk] = load_a(arow, kk * 32, half);

  float best[KNN];
  int bidx[KNN];
#pragma unroll
  for (int k = 0; k < KNN; k++) { best[k] = 3.0e38f; bidx[k] = 0; }

  for (int c = 0; c < 32; c++) {
    int cglob0 = b * N_PTS + c * 128;
    __syncthreads();                       // prev wmma+scan done; reload sC/cs
#if HAVE_TDM
    if (wave == 0)
      tdm_load_bf16_tile((unsigned)(unsigned long long)(void*)sC,
                         (unsigned long long)(const void*)(hbf + (size_t)cglob0 * DIMF),
                         128);
    if (tid < 128) cs[tid] = sq[cglob0 + tid];
    if (wave == 0) __builtin_amdgcn_s_wait_tensorcnt(0);
#else
    {
      const uint4* gc = (const uint4*)(hbf + (size_t)cglob0 * DIMF);
      for (int i = tid; i < 128 * 16; i += 256) {
        int r = i >> 4, ch = i & 15;
        *(uint4*)(sC + r * LDST + ch * 8) = gc[i];
      }
    }
    if (tid < 128) cs[tid] = sq[cglob0 + tid];
#endif
    __syncthreads();

    // 64x128 gram tile: 32 wmma tiles, 4 per wave; store col-major padded
#pragma unroll
    for (int t = 0; t < 4; t++) {
      int ctile = (wave & 1) * 4 + t;
      const unsigned short* brow = &sC[(ctile * 16 + m_in) * LDST];
      v8f acc = {0.f, 0.f, 0.f, 0.f, 0.f, 0.f, 0.f, 0.f};
#pragma unroll
      for (int kk = 0; kk < 4; kk++) {
        v16bf bv = load_b(brow, kk * 32, half);
        acc = __builtin_amdgcn_wmma_f32_16x16x32_bf16(false, av[kk], false, bv,
                                                      (short)0, acc, false, false);
      }
      int col = ctile * 16 + m_in;
      union { v8f v; float4 f[2]; } u;
      u.v = acc;
      float* dcol = sD + col * DSTR + rt * 16 + half * 8;
      *(float4*)(dcol) = u.f[0];
      *(float4*)(dcol + 4) = u.f[1];
    }
    __syncthreads();

    // top-8 scan: rank by cs[j] - 2*gram (query norm is a constant offset)
    if (tid < 64) {
      int qlocal = qlocal0 + tid;
      for (int j = 0; j < 128; j++) {
        int cand = c * 128 + j;
        if (cand == qlocal) continue;      // exclude self
        float d = cs[j] - 2.0f * sD[j * DSTR + tid];
        if (d < best[KNN - 1]) {
          int p = KNN - 1;
          while (p > 0 && best[p - 1] > d) {
            best[p] = best[p - 1]; bidx[p] = bidx[p - 1]; --p;
          }
          best[p] = d; bidx[p] = cand;
        }
      }
    }
  }

  if (tid < 64) {
    int m = qglob0 + tid;
#pragma unroll
    for (int k = 0; k < KNN; k++) idx[m * KNN + k] = b * N_PTS + bidx[k];
  }
}

// ---------------------------------------------------------------------------
// Y[m][n] = sum_k X[m][k] * W[k][n];  Wt is bf16 [n][k]. 64 rows per block.
// LDS: sX 64x136 bf16 | sW 128x136 bf16
__global__ __launch_bounds__(256) void matmul_kernel(const float* __restrict__ X,
                                                     const unsigned short* __restrict__ Wt,
                                                     float* __restrict__ Y) {
  extern __shared__ __align__(16) char smem[];
  unsigned short* sX = (unsigned short*)smem;
  unsigned short* sW = sX + 64 * LDST;

  int row0 = blockIdx.x * 64;
  int tid = threadIdx.x, lane = tid & 31, wave = tid >> 5;
  int half = lane >> 4, m_in = lane & 15;

#if HAVE_TDM
  if (wave == 0)
    tdm_load_bf16_tile((unsigned)(unsigned long long)(void*)sW,
                       (unsigned long long)(const void*)Wt, 128);
#else
  {
    const uint4* gw = (const uint4*)Wt;
    for (int i = tid; i < 128 * 16; i += 256) {
      int r = i >> 4, ch = i & 15;
      *(uint4*)(sW + r * LDST + ch * 8) = gw[i];
    }
  }
#endif
  {
    const float4* gx = (const float4*)(X + (size_t)row0 * DIMF);
    for (int i = tid; i < 64 * 32; i += 256) {
      int r = i >> 5, ch = i & 31;
      float4 v = gx[i];
      uint2 pk;
      pk.x = (unsigned)f2bf(v.x) | ((unsigned)f2bf(v.y) << 16);
      pk.y = (unsigned)f2bf(v.z) | ((unsigned)f2bf(v.w) << 16);
      *(uint2*)(sX + r * LDST + ch * 4) = pk;
    }
  }
#if HAVE_TDM
  if (wave == 0) __builtin_amdgcn_s_wait_tensorcnt(0);
#endif
  __syncthreads();

  int rt = wave >> 1;
  const unsigned short* arow = &sX[(rt * 16 + m_in) * LDST];
  v16bf av[4];
#pragma unroll
  for (int kk = 0; kk < 4; kk++) av[kk] = load_a(arow, kk * 32, half);

#pragma unroll
  for (int t = 0; t < 4; t++) {
    int ctile = (wave & 1) * 4 + t;
    const unsigned short* brow = &sW[(ctile * 16 + m_in) * LDST];
    v8f acc = {0.f, 0.f, 0.f, 0.f, 0.f, 0.f, 0.f, 0.f};
#pragma unroll
    for (int kk = 0; kk < 4; kk++) {
      v16bf bv = load_b(brow, kk * 32, half);
      acc = __builtin_amdgcn_wmma_f32_16x16x32_bf16(false, av[kk], false, bv,
                                                    (short)0, acc, false, false);
    }
    int col = ctile * 16 + m_in;
#pragma unroll
    for (int i = 0; i < 8; i++) {
      int row = rt * 16 + i + half * 8;
      Y[(size_t)(row0 + row) * DIMF + col] = acc[i];
    }
  }
}

// ---------------------------------------------------------------------------
// GCN aggregate: deg is constant 9 (every dst has exactly K=8 in-edges + self).
// out[m] = (sum_{k} hw[idx[m][k]] + hw[m]) / 9 + bias  [+relu | +identity]
__global__ __launch_bounds__(256) void aggregate_kernel(const float* __restrict__ hw,
                                                        const int* __restrict__ idx,
                                                        const float* __restrict__ bias,
                                                        const float* __restrict__ ident,
                                                        float* __restrict__ out,
                                                        int mode) {  // 0=relu, 1=residual
  int m = blockIdx.x * 8 + (threadIdx.x >> 5);
  int lane = threadIdx.x & 31;
  float4 a = ((const float4*)(hw + (size_t)m * DIMF))[lane];
#pragma unroll
  for (int k = 0; k < KNN; k++) {
    int s = idx[m * KNN + k];
    float4 v = ((const float4*)(hw + (size_t)s * DIMF))[lane];
    a.x += v.x; a.y += v.y; a.z += v.z; a.w += v.w;
  }
  const float c = 1.0f / 9.0f;
  float4 bv = ((const float4*)bias)[lane];
  a.x = a.x * c + bv.x; a.y = a.y * c + bv.y;
  a.z = a.z * c + bv.z; a.w = a.w * c + bv.w;
  if (mode == 0) {
    a.x = fmaxf(a.x, 0.f); a.y = fmaxf(a.y, 0.f);
    a.z = fmaxf(a.z, 0.f); a.w = fmaxf(a.w, 0.f);
  } else {
    float4 iv = ((const float4*)(ident + (size_t)m * DIMF))[lane];
    a.x += iv.x; a.y += iv.y; a.z += iv.z; a.w += iv.w;
  }
  ((float4*)(out + (size_t)m * DIMF))[lane] = a;
}

// ---------------------------------------------------------------------------
// Head: z(+bp1) -> LayerNorm -> ReLU -> @Wp2 + bp2 -> tanh*0.1 -> scatter xyz.
__global__ __launch_bounds__(256) void head_kernel(const float* __restrict__ z,
                                                   const float* __restrict__ bp1,
                                                   const float* __restrict__ ln_g,
                                                   const float* __restrict__ ln_b,
                                                   const float* __restrict__ Wp2,
                                                   const float* __restrict__ bp2,
                                                   const float* __restrict__ xyz,
                                                   float* __restrict__ oxyz) {
  int m = blockIdx.x * 8 + (threadIdx.x >> 5);
  int lane = threadIdx.x & 31;
  float4 zv = ((const float4*)(z + (size_t)m * DIMF))[lane];
  float4 b1 = ((const float4*)bp1)[lane];
  zv.x += b1.x; zv.y += b1.y; zv.z += b1.z; zv.w += b1.w;
  float s = zv.x + zv.y + zv.z + zv.w;
  float s2 = zv.x * zv.x + zv.y * zv.y + zv.z * zv.z + zv.w * zv.w;
  for (int off = 16; off; off >>= 1) {
    s += __shfl_xor(s, off, 32);
    s2 += __shfl_xor(s2, off, 32);
  }
  float mean = s * (1.0f / 128.0f);
  float var = s2 * (1.0f / 128.0f) - mean * mean;
  float rstd = rsqrtf(var + 1e-5f);
  float4 gv = ((const float4*)ln_g)[lane];
  float4 bev = ((const float4*)ln_b)[lane];
  float zn[4];
  zn[0] = fmaxf((zv.x - mean) * rstd * gv.x + bev.x, 0.f);
  zn[1] = fmaxf((zv.y - mean) * rstd * gv.y + bev.y, 0.f);
  zn[2] = fmaxf((zv.z - mean) * rstd * gv.z + bev.z, 0.f);
  zn[3] = fmaxf((zv.w - mean) * rstd * gv.w + bev.w, 0.f);
  float p[6];
#pragma unroll
  for (int j = 0; j < 6; j++) {
    int d0 = lane * 4;
    p[j] = zn[0] * Wp2[(d0 + 0) * 6 + j] + zn[1] * Wp2[(d0 + 1) * 6 + j] +
           zn[2] * Wp2[(d0 + 2) * 6 + j] + zn[3] * Wp2[(d0 + 3) * 6 + j];
    for (int off = 16; off; off >>= 1) p[j] += __shfl_xor(p[j], off, 32);
  }
  if (lane == 0) {
    int b = m >> 12, n = m & 4095;
#pragma unroll
    for (int cd = 0; cd < 3; cd++)
#pragma unroll
      for (int u = 0; u < 2; u++) {
        float off6 = tanhf(p[cd * 2 + u] + bp2[cd * 2 + u]) * 0.1f;
        oxyz[b * 24576 + cd * 8192 + u * 4096 + n] =
            xyz[b * 12288 + cd * 4096 + n] + off6;
      }
  }
}

// ---------------------------------------------------------------------------
extern "C" void kernel_launch(void* const* d_in, const int* in_sizes, int n_in,
                              void* d_out, int out_size, void* d_ws, size_t ws_size,
                              hipStream_t stream) {
  (void)in_sizes; (void)n_in; (void)out_size; (void)ws_size;
  const float* xyz  = (const float*)d_in[0];
  const float* feat = (const float*)d_in[1];
  const float* Wa   = (const float*)d_in[2];
  const float* ba   = (const float*)d_in[3];
  const float* Wb   = (const float*)d_in[4];
  const float* bb   = (const float*)d_in[5];
  const float* Wp1  = (const float*)d_in[6];
  const float* bp1  = (const float*)d_in[7];
  const float* ln_g = (const float*)d_in[8];
  const float* ln_b = (const float*)d_in[9];
  const float* Wp2  = (const float*)d_in[10];
  const float* bp2  = (const float*)d_in[11];
  float* out = (float*)d_out;

  // workspace layout (all offsets 256B aligned); total = 30,769,152 bytes
  char* base = (char*)d_ws;
  float*          h    = (float*)(base);                         //  8 MB
  float*          tmp  = (float*)(base + 8388608);               //  8 MB
  float*          hw   = (float*)(base + 16777216);              //  8 MB
  unsigned short* hbf  = (unsigned short*)(base + 25165824);     //  4 MB
  float*          sq   = (float*)(base + 29360128);              // 64 KB
  int*            idx  = (int*)(base + 29425664);                // 512 KB
  unsigned short* Wta  = (unsigned short*)(base + 29949952);     // 384 KB
  unsigned short* Wtb  = (unsigned short*)(base + 30343168);     // 384 KB
  unsigned short* Wtp1 = (unsigned short*)(base + 30736384);     // 32 KB

  const size_t knn_lds = (size_t)(64 * LDST + 128 * LDST) * 2 +
                         (size_t)(128 * DSTR + 128) * 4;         // 87,552 B
  const size_t mm_lds  = (size_t)(64 * LDST + 128 * LDST) * 2;   // 52,224 B

  conv_w_kernel<<<(NBLK * 16384 + 16384 + 255) / 256, 256, 0, stream>>>(
      Wa, Wb, Wp1, Wta, Wtb, Wtp1);
  t_in_kernel<<<(BATCH * DIMF * N_PTS) / 256, 256, 0, stream>>>(feat, h);

  for (int i = 0; i < NBLK; i++) {
    prep_kernel<<<MROWS / 8, 256, 0, stream>>>(h, hbf, sq);
    knn_kernel<<<BATCH * (N_PTS / 64), 256, knn_lds, stream>>>(hbf, sq, idx);
    // GCN 1: relu(gcn(h))
    matmul_kernel<<<MROWS / 64, 256, mm_lds, stream>>>(h, Wta + i * 16384, hw);
    aggregate_kernel<<<MROWS / 8, 256, 0, stream>>>(hw, idx, ba + i * DIMF,
                                                    nullptr, tmp, 0);
    // GCN 2: gcn(tmp) + identity(h) -> h
    matmul_kernel<<<MROWS / 64, 256, mm_lds, stream>>>(tmp, Wtb + i * 16384, hw);
    aggregate_kernel<<<MROWS / 8, 256, 0, stream>>>(hw, idx, bb + i * DIMF,
                                                    h, h, 1);
  }

  // head: z = h @ Wp1 (bias folded into head_kernel)
  matmul_kernel<<<MROWS / 64, 256, mm_lds, stream>>>(h, Wtp1, hw);
  head_kernel<<<MROWS / 8, 256, 0, stream>>>(hw, bp1, ln_g, ln_b, Wp2, bp2,
                                             xyz, out);
  // new_feat at offset B*3*2N = 98304 floats
  t_out_kernel<<<(BATCH * DIMF * N_PTS) / 256, 256, 0, stream>>>(h, out + 98304);
}
